// FasterRCNN_40389872452214
// MI455X (gfx1250) — compile-verified, matrix operand order checked by reading, not run on hardware
//
#include <hip/hip_runtime.h>
#include <hip/hip_bf16.h>
#include <cstddef>

typedef _Float16 f16;
typedef __attribute__((ext_vector_type(16))) _Float16 v16h;
typedef __attribute__((ext_vector_type(8)))  _Float16 v8h;
typedef __attribute__((ext_vector_type(8)))  float    v8f;

static __device__ __forceinline__ int imax(int a,int b){return a>b?a:b;}
static __device__ __forceinline__ int imin(int a,int b){return a<b?a:b;}

static __device__ __forceinline__ v8h zero8h(){
  v8h z;
  #pragma unroll
  for (int e=0;e<8;++e) z[e]=(f16)0.f;
  return z;
}

// ---------------------------------------------------------------- cast f32->f16
__global__ void k_cast_f16(const float* __restrict__ s, f16* __restrict__ d, long n){
  long i = (long)blockIdx.x*blockDim.x + threadIdx.x;
  if (i < n) d[i] = (f16)s[i];
}

// ---------------------------------------------------------------- BN fold: scale/shift
__global__ void k_bnprep(const float* __restrict__ beta, const float* __restrict__ gamma,
                         const float* __restrict__ mean, const float* __restrict__ var,
                         float* __restrict__ scale, float* __restrict__ shift, int C){
  int c = blockIdx.x*blockDim.x + threadIdx.x;
  if (c < C){ float s = gamma[c]*rsqrtf(var[c]+1e-5f); scale[c]=s; shift[c]=beta[c]-mean[c]*s; }
}

// ---------------------------------------------------------------- pack OIHW f32 weights into
// per-lane WMMA A-fragment layout: P[mt][kt][lane(32)][e(16)], K order = (ky*KW+kx)*Cin+cin
template<int KH,int KW>
__global__ void k_packw(const float* __restrict__ W, f16* __restrict__ P,
                        int M,int cinsh,int KT,int MT){
  long i = (long)blockIdx.x*blockDim.x + threadIdx.x;
  long total = (long)MT*KT*512;
  if (i >= total) return;
  const int Cin = 1<<cinsh;
  int e = (int)(i & 15); int L = (int)((i>>4)&31); long t = i>>9;
  int kt = (int)(t % KT); int mt = (int)(t / KT);
  int m = mt*16 + (L & 15);
  int base = (L >= 16) ? 8 : 0;
  int kk = (e < 8) ? (base + e) : (base + 8 + e);   // K within 32-step
  int k = kt*32 + kk;
  int K = (KH*KW) << cinsh;
  float v = 0.f;
  if (m < M && k < K){
    int ci = k & (Cin-1); int t2 = k >> cinsh; int kx = t2 % KW; int ky = t2 / KW;
    v = W[(((size_t)m*Cin + ci)*KH + ky)*KW + kx];
  }
  P[i] = (f16)v;
}

// ---------------------------------------------------------------- B-fragment 8-half run gather
// n already clamped to a valid pixel; garbage columns are never stored.
template<int KH,int KW>
static __device__ __forceinline__ v8h loadBrun(const f16* __restrict__ act,
    int k0,int K,int b,int oy,int ox,int IH,int IW,int cinsh,int stride,int pad){
  const int Cin = 1<<cinsh;
  if (KH==1 && KW==1){
    // 1x1 convs: pad==0, K==Cin (mult of 32) -> unconditional aligned vector load
    return *(const v8h*)(act + ((size_t)(b*IH + oy*stride)*IW + ox*stride)*Cin + k0);
  }
  if (KH==3 && KW==3){
    // 3x3 convs: Cin>=64 pow2 -> run never crosses a tap, never crosses K
    int tp = k0 >> cinsh;             // tap 0..8
    int cin0 = k0 & (Cin-1);
    int kx = tp % 3, ky = tp / 3;
    int iy = oy*stride - pad + ky, ix = ox*stride - pad + kx;
    if (iy >= 0 && iy < IH && ix >= 0 && ix < IW)
      return *(const v8h*)(act + ((size_t)(b*IH + iy)*IW + ix)*Cin + cin0);
    return zero8h();
  }
  // generic (7x7 conv1, Cin==1): elementwise with zero padding + K tail
  v8h o;
  #pragma unroll
  for (int e = 0; e < 8; ++e){
    int k = k0 + e; f16 v = (f16)0.f;
    if (k < K){
      int ci = k & (Cin-1); int t2 = k >> cinsh; int kx2 = t2 % KW; int ky2 = t2 / KW;
      int iy2 = oy*stride - pad + ky2; int ix2 = ox*stride - pad + kx2;
      if (iy2 >= 0 && iy2 < IH && ix2 >= 0 && ix2 < IW)
        v = act[((size_t)(b*IH + iy2)*IW + ix2)*Cin + ci];
    }
    o[e] = v;
  }
  return o;
}

template<int KH,int KW>
static __device__ __forceinline__ v16h loadBfrag(const f16* __restrict__ act,
    int kt,int kbase,int K,int b,int oy,int ox,int IH,int IW,int cinsh,int stride,int pad){
  union { v16h v; v8h h[2]; } f;
  f.h[0] = loadBrun<KH,KW>(act, kt*32 + kbase,      K,b,oy,ox,IH,IW,cinsh,stride,pad);
  f.h[1] = loadBrun<KH,KW>(act, kt*32 + kbase + 16, K,b,oy,ox,IH,IW,cinsh,stride,pad);
  return f.v;
}

// ---------------------------------------------------------------- implicit-GEMM conv (WMMA)
// act NHWC f16; out16 NHWC f16; out32 NCHW f32.
// Wave computes a 32(M) x 32(N) tile: 2 A frags x 2 B frags -> 4 WMMAs (32B loaded / WMMA).
// Software-pipelined 2-deep: loads of kt+1 overlap WMMAs of kt.
template<int KH,int KW>
__global__ __launch_bounds__(256) void k_conv_wmma(
    const f16* __restrict__ Apack, const f16* __restrict__ act,
    f16* __restrict__ out16, float* __restrict__ out32,
    const float* __restrict__ scale, const float* __restrict__ shift,
    const f16* __restrict__ resid,
    int Npix,int IH,int IW,int cinsh,int owsh,int ohsh,
    int M,int MT16,int stride,int pad,int KT,int NW,int relu)
{
  int wave = threadIdx.x >> 5, lane = threadIdx.x & 31;
  int wt = blockIdx.x*8 + wave;
  if (wt >= NW) return;                 // wave-uniform: EXEC all-ones at WMMA
  int mt0 = blockIdx.y*2;
  int mt1 = imin(mt0+1, MT16-1);        // clamp (M<32 heads); dup results discarded by m<M guard
  const int OW = 1<<owsh, OH = 1<<ohsh;
  const int K = (KH*KW) << cinsh;
  int lidx = lane & 15;
  int kbase = (lane >> 4) << 3;
  int n0 = wt*32 + lidx, n1 = n0 + 16;
  int nc0 = imin(n0, Npix-1), nc1 = imin(n1, Npix-1);   // clamp; garbage cols never stored
  int ox0 = nc0 & (OW-1), oy0 = (nc0 >> owsh) & (OH-1), b0 = nc0 >> (owsh+ohsh);
  int ox1 = nc1 & (OW-1), oy1 = (nc1 >> owsh) & (OH-1), b1 = nc1 >> (owsh+ohsh);
  v8f a00 = {}, a01 = {}, a10 = {}, a11 = {};
  const f16* apB0 = Apack + (size_t)mt0*KT*512 + (size_t)lane*16;
  const f16* apB1 = Apack + (size_t)mt1*KT*512 + (size_t)lane*16;
  v16h a0C = *(const v16h*)apB0;
  v16h a1C = *(const v16h*)apB1;
  v16h b0C = loadBfrag<KH,KW>(act,0,kbase,K,b0,oy0,ox0,IH,IW,cinsh,stride,pad);
  v16h b1C = loadBfrag<KH,KW>(act,0,kbase,K,b1,oy1,ox1,IH,IW,cinsh,stride,pad);
  for (int kt = 0; kt < KT; ++kt){
    int ktn = (kt+1 < KT) ? kt+1 : kt;                  // branchless, wave-uniform
    v16h a0N = *(const v16h*)(apB0 + (size_t)ktn*512);
    v16h a1N = *(const v16h*)(apB1 + (size_t)ktn*512);
    __builtin_prefetch((const void*)(apB0 + (size_t)ktn*512 + 4096), 0, 1);
    v16h b0N = loadBfrag<KH,KW>(act,ktn,kbase,K,b0,oy0,ox0,IH,IW,cinsh,stride,pad);
    v16h b1N = loadBfrag<KH,KW>(act,ktn,kbase,K,b1,oy1,ox1,IH,IW,cinsh,stride,pad);
    a00 = __builtin_amdgcn_wmma_f32_16x16x32_f16(false, a0C, false, b0C, (short)0, a00, false, false);
    a01 = __builtin_amdgcn_wmma_f32_16x16x32_f16(false, a0C, false, b1C, (short)0, a01, false, false);
    a10 = __builtin_amdgcn_wmma_f32_16x16x32_f16(false, a1C, false, b0C, (short)0, a10, false, false);
    a11 = __builtin_amdgcn_wmma_f32_16x16x32_f16(false, a1C, false, b1C, (short)0, a11, false, false);
    a0C = a0N; a1C = a1N; b0C = b0N; b1C = b1N;
  }
  #pragma unroll
  for (int mtile = 0; mtile < 2; ++mtile){
    int mb = (mtile ? mt1 : mt0)*16 + ((lane >> 4) << 3);
    #pragma unroll
    for (int tile = 0; tile < 2; ++tile){
      int n = tile ? n1 : n0;
      if (n >= Npix) continue;
      int b = tile ? b1 : b0, oy = tile ? oy1 : oy0, ox = tile ? ox1 : ox0;
      float vals[8];
      #pragma unroll
      for (int r = 0; r < 8; ++r){
        float v = mtile ? (tile ? a11[r] : a10[r]) : (tile ? a01[r] : a00[r]);
        int m = mb + r;
        if (scale && m < M) v *= scale[m];
        if (shift && m < M) v += shift[m];
        vals[r] = v;
      }
      if (resid){                       // resid convs always have M%16==0 -> vector load
        v8h rv = *(const v8h*)(resid + (size_t)n*M + mb);
        #pragma unroll
        for (int r = 0; r < 8; ++r) vals[r] += (float)rv[r];
      }
      if (relu){
        #pragma unroll
        for (int r = 0; r < 8; ++r) vals[r] = vals[r] > 0.f ? vals[r] : 0.f;
      }
      if (out16){                       // f16-producing convs always have M%16==0 -> vector store
        v8h ov;
        #pragma unroll
        for (int r = 0; r < 8; ++r) ov[r] = (f16)vals[r];
        *(v8h*)(out16 + (size_t)n*M + mb) = ov;
      }
      if (out32){                       // strided NCHW f32 (RPN heads, small M)
        #pragma unroll
        for (int r = 0; r < 8; ++r){
          int m = mb + r;
          if (m < M) out32[(((size_t)b*M + m)*OH + oy)*OW + ox] = vals[r];
        }
      }
    }
  }
}

// ---------------------------------------------------------------- GEMM: out[M,N] = A[M,K]*B[N,K]^T + bias[n]
// Requires K%32==0 (true for all uses). 32x32 wave tile, rows clamped.
static __device__ __forceinline__ v16h loadRowFrag(const f16* __restrict__ R, int row, int K,
                                                   int kt, int kbase){
  union { v16h v; v8h h[2]; } f;
  const f16* p = R + (size_t)row*K + kt*32 + kbase;
  f.h[0] = *(const v8h*)p;
  f.h[1] = *(const v8h*)(p + 16);
  return f.v;
}

__global__ __launch_bounds__(256) void k_gemm_wmma(
    const f16* __restrict__ A, const f16* __restrict__ Bm,
    f16* __restrict__ out16, float* __restrict__ out32, const float* __restrict__ bias,
    int M,int N,int K,int KT,int mtsh,int TW,int relu)
{
  int wave = threadIdx.x >> 5, lane = threadIdx.x & 31;
  int id = blockIdx.x*8 + wave;
  if (id >= TW) return;
  int MTm = (1 << mtsh) - 1;
  int mt2 = id & MTm, ntp = id >> mtsh;
  int lidx = lane & 15;
  int ma0 = imin(mt2*32 + lidx,      M-1);
  int ma1 = imin(mt2*32 + 16 + lidx, M-1);
  int n0  = ntp*32 + lidx, n1 = n0 + 16;
  int n0c = imin(n0, N-1), n1c = imin(n1, N-1);
  int kbase = (lane >> 4) << 3;
  v8f a00 = {}, a01 = {}, a10 = {}, a11 = {};
  v16h a0C = loadRowFrag(A,  ma0, K, 0, kbase);
  v16h a1C = loadRowFrag(A,  ma1, K, 0, kbase);
  v16h b0C = loadRowFrag(Bm, n0c, K, 0, kbase);
  v16h b1C = loadRowFrag(Bm, n1c, K, 0, kbase);
  for (int kt = 0; kt < KT; ++kt){
    int ktn = (kt+1 < KT) ? kt+1 : kt;
    v16h a0N = loadRowFrag(A,  ma0, K, ktn, kbase);
    v16h a1N = loadRowFrag(A,  ma1, K, ktn, kbase);
    v16h b0N = loadRowFrag(Bm, n0c, K, ktn, kbase);
    v16h b1N = loadRowFrag(Bm, n1c, K, ktn, kbase);
    a00 = __builtin_amdgcn_wmma_f32_16x16x32_f16(false, a0C, false, b0C, (short)0, a00, false, false);
    a01 = __builtin_amdgcn_wmma_f32_16x16x32_f16(false, a0C, false, b1C, (short)0, a01, false, false);
    a10 = __builtin_amdgcn_wmma_f32_16x16x32_f16(false, a1C, false, b0C, (short)0, a10, false, false);
    a11 = __builtin_amdgcn_wmma_f32_16x16x32_f16(false, a1C, false, b1C, (short)0, a11, false, false);
    a0C = a0N; a1C = a1N; b0C = b0N; b1C = b1N;
  }
  #pragma unroll
  for (int mtile = 0; mtile < 2; ++mtile){
    #pragma unroll
    for (int tile = 0; tile < 2; ++tile){
      int n = tile ? n1 : n0;
      if (n >= N) continue;
      #pragma unroll
      for (int r = 0; r < 8; ++r){
        int m = mt2*32 + mtile*16 + r + ((lane >> 4) << 3);
        if (m >= M) continue;
        float v = (mtile ? (tile ? a11[r] : a10[r]) : (tile ? a01[r] : a00[r]))
                + (bias ? bias[n] : 0.f);
        if (relu) v = v > 0.f ? v : 0.f;
        if (out16) out16[(size_t)m*N + n] = (f16)v;
        if (out32) out32[(size_t)m*N + n] = v;
      }
    }
  }
}

// ---------------------------------------------------------------- maxpool 3x3 s2 p1, NHWC f16
__global__ void k_maxpool(const f16* __restrict__ in, f16* __restrict__ out,
                          int Bn,int IH,int IW,int C){
  int OH = IH/2, OW = IW/2;
  long i = (long)blockIdx.x*blockDim.x + threadIdx.x;
  long total = (long)Bn*OH*OW*C;
  if (i >= total) return;
  int c = (int)(i % C); long t = i / C;
  int ox = (int)(t % OW); t /= OW; int oy = (int)(t % OH); int b = (int)(t / OH);
  float m = -3.4e38f;
  for (int dy = 0; dy < 3; ++dy){ int iy = oy*2 - 1 + dy; if (iy < 0 || iy >= IH) continue;
    for (int dx = 0; dx < 3; ++dx){ int ix = ox*2 - 1 + dx; if (ix < 0 || ix >= IW) continue;
      float v = (float)in[((size_t)(b*IH + iy)*IW + ix)*C + c];
      m = fmaxf(m, v);
    }
  }
  out[i] = (f16)m;
}

// ---------------------------------------------------------------- FPN: dst += nearest-up2(src), NHWC f16
__global__ void k_up2add(f16* __restrict__ dst, const f16* __restrict__ src,
                         int Bn,int H,int W,int C){
  long i = (long)blockIdx.x*blockDim.x + threadIdx.x;
  long total = (long)Bn*H*W*C;
  if (i >= total) return;
  int c = (int)(i % C); long t = i / C;
  int x = (int)(t % W); t /= W; int y = (int)(t % H); int b = (int)(t / H);
  float v = (float)dst[i] + (float)src[(((size_t)b*(H/2) + (y>>1))*(W/2) + (x>>1))*C + c];
  dst[i] = (f16)v;
}

// ---------------------------------------------------------------- ROI pool (torchvision semantics)
__global__ __launch_bounds__(256) void k_roipool(const f16* __restrict__ feat,
                                                 const float* __restrict__ rois,
                                                 f16* __restrict__ rf){
  int i = blockIdx.x;   // roi
  int c = threadIdx.x;  // channel
  const float* r = rois + i*5;
  int b  = (int)r[0];
  int x1 = (int)rintf(r[1]*0.25f), y1 = (int)rintf(r[2]*0.25f);
  int x2 = (int)rintf(r[3]*0.25f), y2 = (int)rintf(r[4]*0.25f);
  int rw = imax(x2 - x1 + 1, 1), rh = imax(y2 - y1 + 1, 1);
  int sx = imin(imax(x1, 0), 108), sy = imin(imax(y1, 0), 108);   // 128 - CROP(20)
  int offx = x1 - sx, offy = y1 - sy;
  float bsx = rw/7.0f, bsy = rh/7.0f;
  for (int py = 0; py < 7; ++py){
    int ys = (int)floorf(py*bsy), ye = imin((int)ceilf((py+1)*bsy), rh);
    int jy0 = imax(ys + offy, 0), jy1 = imin(ye + offy, 20);
    for (int px = 0; px < 7; ++px){
      int xs = (int)floorf(px*bsx), xe = imin((int)ceilf((px+1)*bsx), rw);
      int jx0 = imax(xs + offx, 0), jx1 = imin(xe + offx, 20);
      float m = -1e30f;
      for (int jy = jy0; jy < jy1; ++jy){ int iy = sy + jy;
        for (int jx = jx0; jx < jx1; ++jx){ int ix = sx + jx;
          m = fmaxf(m, (float)feat[((size_t)(b*128 + iy)*128 + ix)*256 + c]);
        }
      }
      if (m <= -5e29f) m = 0.f;  // empty bin -> 0
      rf[(size_t)i*12544 + (size_t)c*49 + py*7 + px] = (f16)m;
    }
  }
}

// ---------------------------------------------------------------- anchors per level
__global__ void k_anchors(float* __restrict__ out, int g, float st, float s){
  int i = blockIdx.x*blockDim.x + threadIdx.x;
  int total = g*g*3;
  if (i >= total) return;
  int a = i % 3; int t = i/3; int iy = t % g; int ix = t / g;   // ix outer (meshgrid 'ij')
  float r = (a == 0) ? 0.5f : ((a == 1) ? 1.0f : 2.0f);
  float h = s*sqrtf(r), w = s/sqrtf(r);
  float off = floorf(st*0.5f);
  float cx = ix*st + off, cy = iy*st + off;
  out[(size_t)i*4+0] = cx - w*0.5f;
  out[(size_t)i*4+1] = cy - h*0.5f;
  out[(size_t)i*4+2] = cx + w*0.5f;
  out[(size_t)i*4+3] = cy + h*0.5f;
}

// =================================================================================
static inline long cdivl(long a, long b){ return (a + b - 1)/b; }
static inline int  ilog2i(int v){ int s=0; while ((1<<s) < v) ++s; return s; }

extern "C" void kernel_launch(void* const* d_in, const int* in_sizes, int n_in,
                              void* d_out, int out_size, void* d_ws, size_t ws_size,
                              hipStream_t strm){
  (void)in_sizes; (void)n_in; (void)out_size; (void)ws_size;
  const float* images = (const float*)d_in[0];
  const float* rois   = (const float*)d_in[1];
  float* out = (float*)d_out;

  // ---- params leaf cursor: jax tree order (sorted dict keys, lists in order)
  int cur = 2;
  auto nx = [&](){ return (const float*)d_in[cur++]; };
  const float *W_bbox=nx(), *b_bbox=nx();
  const float *bn1p[4]; for (int q=0;q<4;++q) bn1p[q]=nx();            // beta,gamma,mean,var
  const float *W_cls=nx(), *b_cls=nx();
  const float *W_c1=nx();
  const float *W_fc1=nx(), *b_fc1=nx();
  const float *W_fc2=nx(), *b_fc2=nx();
  const float *fiW[4], *fiB[4]; for (int i=0;i<4;++i){ fiW[i]=nx(); fiB[i]=nx(); }
  const float *flW[4], *flB[4]; for (int i=0;i<4;++i){ flW[i]=nx(); flB[i]=nx(); }
  const float *p6W=nx(), *p6B=nx();
  struct Blk { const float* bn[4][4]; const float *w1,*w2,*w3,*wd; };
  static const int counts[4] = {3,4,6,3};
  Blk blocks[4][6];
  for (int li=0; li<4; ++li)
    for (int bi=0; bi<counts[li]; ++bi){
      Blk& B_ = blocks[li][bi];
      for (int j=0;j<3;++j) for (int q=0;q<4;++q) B_.bn[j][q]=nx();     // bn1,bn2,bn3
      if (bi==0) for (int q=0;q<4;++q) B_.bn[3][q]=nx();                // bnd
      B_.w1=nx(); B_.w2=nx(); B_.w3=nx();
      B_.wd = (bi==0) ? nx() : nullptr;
    }
  const float *rbW=nx(), *rbB=nx();    // rpn_bbox
  const float *rcW=nx(), *rcB=nx();    // rpn_cls
  const float *rvW=nx(), *rvB=nx();    // rpn_conv

  // ---- workspace bump allocator
  char* base = (char*)d_ws; size_t off = 0;
  auto alloc = [&](size_t bytes)->void*{ size_t o=(off+255)&~(size_t)255; off=o+bytes; return base+o; };
  f16*  wpack = (f16*)alloc((size_t)16<<20);          // shared conv weight pack scratch
  float* ssb  = (float*)alloc(2*2048*sizeof(float));  // BN scale||shift scratch
  f16*  S0 = (f16*)alloc((size_t)17<<20);
  f16*  S1 = (f16*)alloc((size_t)17<<20);
  f16*  S2 = (f16*)alloc((size_t)17<<20);
  f16*  M1 = (f16*)alloc((size_t)9<<20);
  f16*  M2 = (f16*)alloc((size_t)9<<20);
  f16*  FCW = (f16*)alloc((size_t)26<<20);
  f16*  img16 = (f16*)alloc((size_t)2*512*512*2);
  f16*  a0  = (f16*)alloc((size_t)2*256*256*64*2);
  f16*  x0  = (f16*)alloc((size_t)2*128*128*64*2);
  f16*  Cs[4];
  Cs[0]=(f16*)alloc((size_t)2*128*128*256*2);
  Cs[1]=(f16*)alloc((size_t)2*64*64*512*2);
  Cs[2]=(f16*)alloc((size_t)2*32*32*1024*2);
  Cs[3]=(f16*)alloc((size_t)2*16*16*2048*2);
  f16* I2=(f16*)alloc((size_t)2*128*128*256*2);
  f16* I3=(f16*)alloc((size_t)2*64*64*256*2);
  f16* I4=(f16*)alloc((size_t)2*32*32*256*2);
  f16* I5=(f16*)alloc((size_t)2*16*16*256*2);
  f16* P2=(f16*)alloc((size_t)2*128*128*256*2);
  f16* P3=(f16*)alloc((size_t)2*64*64*256*2);
  f16* P4=(f16*)alloc((size_t)2*32*32*256*2);
  f16* P5=(f16*)alloc((size_t)2*16*16*256*2);
  f16* P6=(f16*)alloc((size_t)2*8*8*256*2);
  f16* rpnPC=(f16*)alloc((size_t)2<<20);              // rpn_conv pack (persists over 5 levels)
  f16* rpnPK=(f16*)alloc((size_t)64<<10);             // rpn_cls pack
  f16* rpnPB=(f16*)alloc((size_t)64<<10);             // rpn_bbox pack
  f16* rf=(f16*)alloc((size_t)256*12544*2);
  f16* h1=(f16*)alloc((size_t)256*1024*2);
  f16* h2=(f16*)alloc((size_t)256*1024*2);

  // ---- launch helpers
  auto cast16 = [&](const float* s, f16* d, long n){
    k_cast_f16<<<dim3((unsigned)cdivl(n,256)),dim3(256),0,strm>>>(s,d,n);
  };
  auto packw = [&](const float* W, f16* dst, int M,int Cin,int KH, int* pKT,int* pMT){
    int K=Cin*KH*KH, KT=(K+31)/32, MT=(M+15)/16, cinsh=ilog2i(Cin);
    long total=(long)MT*KT*512;
    dim3 g((unsigned)cdivl(total,256));
    if (KH==1)      k_packw<1,1><<<g,dim3(256),0,strm>>>(W,dst,M,cinsh,KT,MT);
    else if (KH==3) k_packw<3,3><<<g,dim3(256),0,strm>>>(W,dst,M,cinsh,KT,MT);
    else            k_packw<7,7><<<g,dim3(256),0,strm>>>(W,dst,M,cinsh,KT,MT);
    *pKT=KT; *pMT=MT;
  };
  auto conv = [&](const f16* pack, const f16* in, f16* o16, float* o32,
                  const float* scale, const float* shift, const f16* resid,
                  int Bn,int IH,int IW,int Cin,int M,int KH,int st,int pad,int relu,
                  int KT,int MT16){
    int OH=(IH+2*pad-KH)/st+1, OW=(IW+2*pad-KH)/st+1;
    int Npix=Bn*OH*OW, NW=(Npix+31)/32;
    int cinsh=ilog2i(Cin), owsh=ilog2i(OW), ohsh=ilog2i(OH);
    dim3 g((unsigned)cdivl(NW,8),(unsigned)((M+31)/32));
    if (KH==1)
      k_conv_wmma<1,1><<<g,dim3(256),0,strm>>>(pack,in,o16,o32,scale,shift,resid,
                                               Npix,IH,IW,cinsh,owsh,ohsh,M,MT16,st,pad,KT,NW,relu);
    else if (KH==3)
      k_conv_wmma<3,3><<<g,dim3(256),0,strm>>>(pack,in,o16,o32,scale,shift,resid,
                                               Npix,IH,IW,cinsh,owsh,ohsh,M,MT16,st,pad,KT,NW,relu);
    else
      k_conv_wmma<7,7><<<g,dim3(256),0,strm>>>(pack,in,o16,o32,scale,shift,resid,
                                               Npix,IH,IW,cinsh,owsh,ohsh,M,MT16,st,pad,KT,NW,relu);
  };
  auto convbn = [&](const float* W, const float* const* bn4, const f16* in, f16* o16,
                    const f16* resid, int Bn,int IH,int IW,int Cin,int M,int KH,
                    int st,int pad,int relu){
    int KT,MT; packw(W,wpack,M,Cin,KH,&KT,&MT);
    k_bnprep<<<dim3((unsigned)cdivl(M,256)),dim3(256),0,strm>>>(bn4[0],bn4[1],bn4[2],bn4[3],ssb,ssb+2048,M);
    conv(wpack,in,o16,nullptr,ssb,ssb+2048,resid,Bn,IH,IW,Cin,M,KH,st,pad,relu,KT,MT);
  };
  auto convb = [&](const float* W, const float* bias, const f16* in, f16* o16, float* o32,
                   int Bn,int IH,int IW,int Cin,int M,int KH,int st,int pad,int relu){
    int KT,MT; packw(W,wpack,M,Cin,KH,&KT,&MT);
    conv(wpack,in,o16,o32,nullptr,bias,nullptr,Bn,IH,IW,Cin,M,KH,st,pad,relu,KT,MT);
  };
  auto gemm = [&](const f16* A, const f16* Bm, f16* o16, float* o32, const float* bias,
                  int M,int N,int K,int relu){
    int KT=(K+31)/32, MT2=(M+31)/32, mtsh=ilog2i(MT2), NW=(N+31)/32, TW=MT2*NW;
    k_gemm_wmma<<<dim3((unsigned)cdivl(TW,8)),dim3(256),0,strm>>>(A,Bm,o16,o32,bias,M,N,K,KT,mtsh,TW,relu);
  };
  auto up2add = [&](f16* dst, const f16* src, int Bn,int H,int W,int C){
    long total=(long)Bn*H*W*C;
    k_up2add<<<dim3((unsigned)cdivl(total,256)),dim3(256),0,strm>>>(dst,src,Bn,H,W,C);
  };

  // ================= stem =================
  cast16(images, img16, 2l*512*512);
  convbn(W_c1, bn1p, img16, a0, nullptr, 2,512,512,1, 64,7,2,3, 1);
  {
    long total = 2l*128*128*64;
    k_maxpool<<<dim3((unsigned)cdivl(total,256)),dim3(256),0,strm>>>(a0,x0,2,256,256,64);
  }

  // ================= ResNet layers =================
  static const int mids_[4]={64,128,256,512}, ocs_[4]={256,512,1024,2048};
  f16* curb = x0; int curC = 64, curH = 128;
  for (int li=0; li<4; ++li){
    int mid = mids_[li], oc = ocs_[li], nb = counts[li];
    for (int bi=0; bi<nb; ++bi){
      Blk& bk = blocks[li][bi];
      int st_ = (li>0 && bi==0) ? 2 : 1;
      int Hin = curH, Hout = Hin/st_;
      convbn(bk.w1, bk.bn[0], curb, M1, nullptr, 2,Hin,Hin,curC, mid,1,1,0, 1);
      convbn(bk.w2, bk.bn[1], M1,   M2, nullptr, 2,Hin,Hin,mid,  mid,3,st_,1, 1);
      const f16* res;
      if (bk.wd){
        convbn(bk.wd, bk.bn[3], curb, S2, nullptr, 2,Hin,Hin,curC, oc,1,st_,0, 0);
        res = S2;
      } else res = curb;
      f16* ob = (bi == nb-1) ? Cs[li] : ((bi & 1) ? S1 : S0);
      convbn(bk.w3, bk.bn[2], M2, ob, res, 2,Hout,Hout,mid, oc,1,1,0, 1);
      curb = ob; curC = oc; curH = Hout;
    }
  }

  // ================= FPN =================
  convb(fiW[3], fiB[3], Cs[3], I5, nullptr, 2,16, 16, 2048, 256,1,1,0, 0);
  convb(fiW[2], fiB[2], Cs[2], I4, nullptr, 2,32, 32, 1024, 256,1,1,0, 0);
  up2add(I4, I5, 2,32,32,256);
  convb(fiW[1], fiB[1], Cs[1], I3, nullptr, 2,64, 64, 512,  256,1,1,0, 0);
  up2add(I3, I4, 2,64,64,256);
  convb(fiW[0], fiB[0], Cs[0], I2, nullptr, 2,128,128,256,  256,1,1,0, 0);
  up2add(I2, I3, 2,128,128,256);
  convb(flW[0], flB[0], I2, P2, nullptr, 2,128,128,256, 256,3,1,1, 0);
  convb(flW[1], flB[1], I3, P3, nullptr, 2,64, 64, 256, 256,3,1,1, 0);
  convb(flW[2], flB[2], I4, P4, nullptr, 2,32, 32, 256, 256,3,1,1, 0);
  convb(flW[3], flB[3], I5, P5, nullptr, 2,16, 16, 256, 256,3,1,1, 0);
  convb(p6W,    p6B,    Cs[3], P6, nullptr, 2,16,16,2048, 256,3,2,1, 0);

  // ================= RPN heads -> d_out (obj/reg, f32 NCHW flatten) =================
  int rvKT,rvMT, rcKT,rcMT, rbKT,rbMT;
  packw(rvW, rpnPC, 256,256,3, &rvKT,&rvMT);
  packw(rcW, rpnPK, 3,  256,1, &rcKT,&rcMT);
  packw(rbW, rpnPB, 12, 256,1, &rbKT,&rbMT);
  f16* feats[5] = {P2,P3,P4,P5,P6};
  int  fh[5]    = {128,64,32,16,8};
  size_t objOff = 6400;      // after cls(1280)+bbox(5120)
  size_t regOff = 137344;    // after obj(130944)
  for (int l=0; l<5; ++l){
    int H = fh[l];
    conv(rpnPC, feats[l], S0, nullptr, nullptr, rvB, nullptr, 2,H,H,256, 256,3,1,1, 1, rvKT,rvMT);
    conv(rpnPK, S0, nullptr, out+objOff, nullptr, rcB, nullptr, 2,H,H,256, 3, 1,1,0, 0, rcKT,rcMT);
    conv(rpnPB, S0, nullptr, out+regOff, nullptr, rbB, nullptr, 2,H,H,256, 12,1,1,0, 0, rbKT,rbMT);
    objOff += (size_t)2*3*H*H;
    regOff += (size_t)2*12*H*H;
  }

  // ================= ROI pool + FC head =================
  k_roipool<<<dim3(256),dim3(256),0,strm>>>(P2, rois, rf);
  cast16(W_fc1, FCW, 1024l*12544);
  gemm(rf, FCW, h1, nullptr, b_fc1, 256,1024,12544, 1);
  cast16(W_fc2, FCW, 1024l*1024);
  gemm(h1, FCW, h2, nullptr, b_fc2, 256,1024,1024, 1);
  cast16(W_cls, FCW, 5l*1024);
  gemm(h2, FCW, nullptr, out + 0,    b_cls,  256,5, 1024, 0);
  cast16(W_bbox, FCW, 20l*1024);
  gemm(h2, FCW, nullptr, out + 1280, b_bbox, 256,20,1024, 0);

  // ================= anchors =================
  {
    size_t aoff = 661120;
    int   g5[5] = {128,64,32,16,8};
    float s5[5] = {32.f,64.f,128.f,256.f,512.f};
    for (int l=0; l<5; ++l){
      int g = g5[l]; float st = 512.f/(float)g;
      long total = (long)g*g*3;
      k_anchors<<<dim3((unsigned)cdivl(total,256)),dim3(256),0,strm>>>(out+aoff, g, st, s5[l]);
      aoff += (size_t)total*4;
    }
  }
}
